// SEB_with_Attn_31808527794655
// MI455X (gfx1250) — matrix operand order repaired and checked by reading, not compiled, not run on hardware
//
#include <hip/hip_runtime.h>

// ---------------------------------------------------------------------------
// MI455X (gfx1250) implementation of Conv+BN+LReLU -> self-attention block.
// All GEMM-like work on v_wmma_f32_16x16x32_f16 (f16 in, f32 accumulate).
// Flash-attention (online softmax): the 537MB attn matrix is never formed.
// Round 3: all register-array loops fully unrolled (constant indices -> VGPRs,
// no scratch); register pressure controlled with scheduling barriers between
// fragment builds instead of partial unrolling.
// ---------------------------------------------------------------------------

typedef __attribute__((ext_vector_type(16))) _Float16 v16h;
typedef __attribute__((ext_vector_type(8)))  float    v8f;

#define B_      8
#define CIN     128
#define COUT    256
#define HIN     128
#define WIN     128
#define NPIX    4096      // 64*64 output pixels
#define DQK     32        // C/8
#define QKVW    320       // 32 q | 32 k | 256 v fused row
#define BN_EPS  1e-5f
#define SLOPE   0.01f

// Stops the scheduler from hoisting every fragment's loads at once (round-1
// spill cause) without forcing register arrays into scratch (round-2 cause).
#define SCHED_BARRIER() asm volatile("" ::: "memory")

static __device__ __forceinline__ v8f wmma16(v16h a, v16h b, v8f c) {
  // (neg_a, A, neg_b, B, c_mod, C, reuse_a, reuse_b)
  return __builtin_amdgcn_wmma_f32_16x16x32_f16(false, a, false, b, (short)0, c,
                                                false, false);
}

// A-matrix (16x32 f16) K index for vgpr-pair v in this lane-half
// (ISA 7.12.2: V0..3 -> K {0..7}+8*half, V4..7 -> K {16..23}+8*half)
static __device__ __forceinline__ int a_k0(int v, int half) {
  return ((v < 4) ? (2 * v) : (16 + 2 * (v - 4))) + 8 * half;
}

// ---------------------------------------------------------------------------
// Kernel 1: 3x3 s2 conv as implicit GEMM (M=pixels, K=Cin*9, N=Cout)
//   + BatchNorm(inference) + LeakyReLU.
// Per wave: 16 pixels x 128 cout (gridDim.z splits the 256 couts in half).
// Writes x (B,C,N) f32 for the residual, and x^T (B,N,C) f16 for the GEMMs.
// ---------------------------------------------------------------------------
__global__ __launch_bounds__(128) void conv_bn_lrelu_kernel(
    const float* __restrict__ feat, const float* __restrict__ conv_w,
    const float* __restrict__ bn_scale, const float* __restrict__ bn_bias,
    const float* __restrict__ bn_mean, const float* __restrict__ bn_var,
    float* __restrict__ x, _Float16* __restrict__ xt) {
  const int lane = threadIdx.x & 31;
  const int wv   = threadIdx.x >> 5;
  const int half = lane >> 4;
  const int lm   = lane & 15;
  const int b    = blockIdx.y;
  const int n0   = blockIdx.x * 64 + wv * 16;   // 16 output pixels per wave
  const int cob  = blockIdx.z * 128;            // cout half

  v8f acc[8];
#pragma unroll
  for (int t = 0; t < 8; ++t) acc[t] = {};

  const int n  = n0 + lm;                       // A-fragment row for this lane
  const int oh = n >> 6, ow = n & 63;
  const int kB = lm + 16 * half;                // unused lane index (0..31)

  for (int kb = 0; kb < 36; ++kb) {             // K = 9 taps * 128 cin (rolled)
    const int kbase = kb * 32;
    const int tap   = kbase >> 7;               // which (kh,kw)
    const int ci0   = kbase & 127;
    const int kh = tap / 3, kw = tap % 3;
    const int ih = 2 * oh - 1 + kh;
    const int iw = 2 * ow - 1 + kw;
    const bool inb = (ih >= 0) && (ih < HIN) && (iw >= 0) && (iw < WIN);
    const int ihc = min(max(ih, 0), HIN - 1);
    const int iwc = min(max(iw, 0), WIN - 1);
    const size_t fbase = (((size_t)b * CIN) * HIN + ihc) * WIN + iwc;

    // A fragment: im2col rows (16 pixels x 32 of K)
    v16h af;
#pragma unroll
    for (int v = 0; v < 8; ++v) {
      const int k0 = a_k0(v, half);
#pragma unroll
      for (int e = 0; e < 2; ++e) {
        const int ci = ci0 + k0 + e;
        const float val = inb ? feat[fbase + (size_t)ci * (HIN * WIN)] : 0.0f;
        af[2 * v + e] = (_Float16)val;
      }
    }
    SCHED_BARRIER();

    // B fragments: W^T (K x 16 cout) per output tile; fully unrolled so
    // acc[ct] stays in VGPRs; barrier per tile caps in-flight loads.
    const float* wp = conv_w + (size_t)ci0 * 9 + tap + (size_t)(cob + lm) * (CIN * 9);
#pragma unroll
    for (int ct = 0; ct < 8; ++ct) {
      v16h bf;
#pragma unroll
      for (int j = 0; j < 16; ++j)
        bf[j] = (_Float16)wp[(size_t)(ct * 16) * (CIN * 9) + (size_t)(a_k0(j >> 1, half) + (j & 1)) * 9];
      acc[ct] = wmma16(af, bf, acc[ct]);
      SCHED_BARRIER();
    }
  }

  // BN + LeakyReLU epilogue; scatter to (C,N) f32 and (N,C) f16
#pragma unroll
  for (int ct = 0; ct < 8; ++ct) {
    const int co = cob + ct * 16 + lm;
    const float inv = bn_scale[co] * rsqrtf(bn_var[co] + BN_EPS);
    const float mu = bn_mean[co];
    const float bi = bn_bias[co];
#pragma unroll
    for (int r = 0; r < 8; ++r) {
      const int nn = n0 + r + 8 * half;         // C/D row = r + 8*half
      float vv = (acc[ct][r] - mu) * inv + bi;
      vv = (vv >= 0.0f) ? vv : SLOPE * vv;
      x[((size_t)b * COUT + co) * NPIX + nn] = vv;
      xt[((size_t)b * NPIX + nn) * COUT + co] = (_Float16)vv;
    }
    SCHED_BARRIER();
  }
}

// ---------------------------------------------------------------------------
// Kernel 2: fused q/k/v 1x1 convs: (N x 256) @ (256 x {32,32,256}) + bias
// Output packed per pixel: qkv[b][n][0:32]=q, [32:64]=k, [64:320]=v  (f16)
// ---------------------------------------------------------------------------
__global__ __launch_bounds__(128) void qkv_kernel(
    const _Float16* __restrict__ xt,
    const float* __restrict__ q_w, const float* __restrict__ q_b,
    const float* __restrict__ k_w, const float* __restrict__ k_b,
    const float* __restrict__ v_w, const float* __restrict__ v_b,
    _Float16* __restrict__ qkv) {
  const int lane = threadIdx.x & 31;
  const int wv   = threadIdx.x >> 5;
  const int half = lane >> 4;
  const int lm   = lane & 15;
  const int b    = blockIdx.y;
  const int n0   = blockIdx.x * 64 + wv * 16;

  // Preload all A fragments (16 rows x full K=256) from row-major x^T
  v16h af[8];
  const _Float16* row = xt + ((size_t)b * NPIX + n0 + lm) * COUT;
#pragma unroll
  for (int kb = 0; kb < 8; ++kb) {
#pragma unroll
    for (int v = 0; v < 8; ++v) {
      const int k0 = a_k0(v, half) + kb * 32;
      af[kb][2 * v]     = row[k0];
      af[kb][2 * v + 1] = row[k0 + 1];
    }
  }

  const int kB = lm + 16 * half;
#pragma unroll 1
  for (int ct = 0; ct < 20; ++ct) {             // 2 q, 2 k, 16 v tiles (rolled)
    const float* wm; const float* bias; int co0, oo;
    if (ct < 2)      { wm = q_w; bias = q_b; co0 = ct * 16;       oo = ct * 16; }
    else if (ct < 4) { wm = k_w; bias = k_b; co0 = (ct - 2) * 16; oo = 32 + (ct - 2) * 16; }
    else             { wm = v_w; bias = v_b; co0 = (ct - 4) * 16; oo = 64 + (ct - 4) * 16; }

    v8f acc = {};
#pragma unroll
    for (int kb = 0; kb < 8; ++kb) {            // fully unrolled: af[kb] const
      const int c = kb * 32;
      v16h bf;
#pragma unroll
      for (int j = 0; j < 16; ++j)
        bf[j] = (_Float16)wm[(size_t)(co0 + lm) * COUT + c + a_k0(j >> 1, half) + (j & 1)];
      acc = wmma16(af[kb], bf, acc);
      SCHED_BARRIER();
    }
    const float bi = bias[co0 + lm];
#pragma unroll
    for (int r = 0; r < 8; ++r) {
      const int nn = n0 + r + 8 * half;
      qkv[((size_t)b * NPIX + nn) * QKVW + oo + lm] = (_Float16)(acc[r] + bi);
    }
  }
}

// ---------------------------------------------------------------------------
// Kernel 3: flash attention + gamma*out + x residual.
// Block = 4 waves = 2 m-tiles x 2 channel-halves. Each wave owns 16 query
// rows and a 16x128 f32 accumulator (8 C/D tiles = 64 VGPRs). The two waves
// of a channel pair redundantly compute S + softmax (S is ~8% of the FLOPs)
// so waves stay fully independent - no cross-wave sync in the loop.
// Per-row softmax stats live in the same (r, lane-half) layout as C
// fragments, so acc rescaling is a straight per-element multiply.
// ---------------------------------------------------------------------------
__global__ __launch_bounds__(128) void flash_attn_kernel(
    const _Float16* __restrict__ qkv, const float* __restrict__ x,
    const float* __restrict__ gamma, float* __restrict__ out) {
  __shared__ _Float16 plds[4 * 16 * 128];       // per-wave 16x128 P tile
  const int lane = threadIdx.x & 31;
  const int wv   = threadIdx.x >> 5;
  const int half = lane >> 4;
  const int lm   = lane & 15;
  const int b    = blockIdx.y;
  const int m0   = blockIdx.x * 32 + (wv >> 1) * 16;  // m-tile of this wave
  const int cob  = (wv & 1) * 128;                    // channel half
  _Float16* P = plds + wv * (16 * 128);

  // q fragment (16 rows x K=32)
  v16h qf;
  const _Float16* qrow = qkv + ((size_t)b * NPIX + m0 + lm) * QKVW;
#pragma unroll
  for (int v = 0; v < 8; ++v) {
    const int k0 = a_k0(v, half);
    qf[2 * v]     = qrow[k0];
    qf[2 * v + 1] = qrow[k0 + 1];
  }

  v8f acc[8];
#pragma unroll
  for (int t = 0; t < 8; ++t) acc[t] = {};
  float rmax[8], lsum[8];
#pragma unroll
  for (int r = 0; r < 8; ++r) { rmax[r] = -1e30f; lsum[r] = 0.0f; }

  const int dB = lm + 16 * half;                // key row offset (0..31) for the k prefetch
  for (int nb = 0; nb < NPIX; nb += 128) {      // 128-key blocks (rolled)
    // Prefetch next key block's k-vectors into cache (global_prefetch_b8)
    if (nb + 128 < NPIX)
      __builtin_prefetch(
          qkv + ((size_t)b * NPIX + nb + 128 + dB) * QKVW + DQK, 0, 1);

    // S tiles: q (16x32) @ k^T (32x16) per 16-column tile (fully unrolled)
    v8f s[8];
#pragma unroll
    for (int t = 0; t < 8; ++t) {
      v16h bf;
#pragma unroll
      for (int j = 0; j < 16; ++j)
        bf[j] = qkv[((size_t)b * NPIX + nb + t * 16 + lm) * QKVW + DQK + a_k0(j >> 1, half) + (j & 1)];
      v8f z = {};
      s[t] = wmma16(qf, bf, z);
      SCHED_BARRIER();
    }

    // Online softmax: row max / rescale / row sum (half-wave shuffles)
    float nm[8], sc[8];
#pragma unroll
    for (int r = 0; r < 8; ++r) {
      float mx = s[0][r];
#pragma unroll
      for (int t = 1; t < 8; ++t) mx = fmaxf(mx, s[t][r]);
      mx = fmaxf(mx, __shfl_xor(mx, 1));
      mx = fmaxf(mx, __shfl_xor(mx, 2));
      mx = fmaxf(mx, __shfl_xor(mx, 4));
      mx = fmaxf(mx, __shfl_xor(mx, 8));
      nm[r]  = fmaxf(rmax[r], mx);
      sc[r]  = __expf(rmax[r] - nm[r]);
      rmax[r] = nm[r];
    }
#pragma unroll
    for (int r = 0; r < 8; ++r) {
      float rs = 0.0f;
#pragma unroll
      for (int t = 0; t < 8; ++t) {
        const float p = __expf(s[t][r] - nm[r]);
        s[t][r] = p;
        rs += p;
      }
      rs += __shfl_xor(rs, 1);
      rs += __shfl_xor(rs, 2);
      rs += __shfl_xor(rs, 4);
      rs += __shfl_xor(rs, 8);
      lsum[r] = lsum[r] * sc[r] + rs;
    }
#pragma unroll
    for (int t = 0; t < 8; ++t)
#pragma unroll
      for (int r = 0; r < 8; ++r) acc[t][r] *= sc[r];

    // Re-shape P: C-fragment layout -> row-major LDS -> A-fragment layout
#pragma unroll
    for (int t = 0; t < 8; ++t)
#pragma unroll
      for (int r = 0; r < 8; ++r)
        P[(r + 8 * half) * 128 + t * 16 + lm] = (_Float16)s[t][r];
    asm volatile("s_wait_dscnt 0" ::: "memory");  // DS in-order; belt+braces

    // acc += P (16x128) @ V (128x128-half), 4 K-steps x 8 channel tiles
#pragma unroll 1
    for (int kb = 0; kb < 4; ++kb) {            // rolled: no array indexing
      v16h pf;
#pragma unroll
      for (int v = 0; v < 8; ++v) {
        const int k0 = a_k0(v, half) + kb * 32;
        pf[2 * v]     = P[lm * 128 + k0];
        pf[2 * v + 1] = P[lm * 128 + k0 + 1];
      }
      const _Float16* vrow =
          qkv + ((size_t)b * NPIX + nb + kb * 32) * QKVW + 64 + cob + lm;
#pragma unroll
      for (int t = 0; t < 8; ++t) {             // fully unrolled: acc[t] const
        v16h bf;
#pragma unroll
        for (int j = 0; j < 16; ++j)
          bf[j] = vrow[(size_t)(a_k0(j >> 1, half) + (j & 1)) * QKVW + t * 16];
        acc[t] = wmma16(pf, bf, acc[t]);
        SCHED_BARRIER();
      }
    }
  }

  // out = gamma * (acc / l) + x
  const float g = gamma[0];
#pragma unroll
  for (int t = 0; t < 8; ++t) {
    const int o = cob + t * 16 + lm;
#pragma unroll
    for (int r = 0; r < 8; ++r) {
      const int mm = m0 + r + 8 * half;
      const size_t idx = ((size_t)b * COUT + o) * NPIX + mm;
      out[idx] = g * (acc[t][r] / lsum[r]) + x[idx];
    }
    SCHED_BARRIER();
  }
}

// ---------------------------------------------------------------------------
extern "C" void kernel_launch(void* const* d_in, const int* in_sizes, int n_in,
                              void* d_out, int out_size, void* d_ws,
                              size_t ws_size, hipStream_t stream) {
  (void)in_sizes; (void)n_in; (void)out_size; (void)ws_size;
  const float* feat     = (const float*)d_in[0];
  const float* conv_w   = (const float*)d_in[1];
  const float* bn_scale = (const float*)d_in[2];
  const float* bn_bias  = (const float*)d_in[3];
  const float* bn_mean  = (const float*)d_in[4];
  const float* bn_var   = (const float*)d_in[5];
  const float* q_w      = (const float*)d_in[6];
  const float* q_b      = (const float*)d_in[7];
  const float* k_w      = (const float*)d_in[8];
  const float* k_b      = (const float*)d_in[9];
  const float* v_w      = (const float*)d_in[10];
  const float* v_b      = (const float*)d_in[11];
  const float* gamma    = (const float*)d_in[12];

  // Workspace: x f32 (33.5MB) | x^T f16 (16.8MB) | qkv f16 (21MB) ~= 71.3MB
  char* ws = (char*)d_ws;
  float*    x   = (float*)ws;
  _Float16* xt  = (_Float16*)(ws + (size_t)33554432);
  _Float16* qkv = (_Float16*)(ws + (size_t)33554432 + 16777216);

  dim3 block(128);
  dim3 grid1(NPIX / 64, B_, 2);          // conv: 16 pix x 128 cout per wave
  dim3 grid2(NPIX / 64, B_);             // qkv
  dim3 grid3(NPIX / 32, B_);             // flash: 2 m-tiles x 2 ch-halves
  conv_bn_lrelu_kernel<<<grid1, block, 0, stream>>>(
      feat, conv_w, bn_scale, bn_bias, bn_mean, bn_var, x, xt);
  qkv_kernel<<<grid2, block, 0, stream>>>(xt, q_w, q_b, k_w, k_b, v_w, v_b, qkv);
  flash_attn_kernel<<<grid3, block, 0, stream>>>(qkv, x, gamma, (float*)d_out);
}
